// Model_28922309771634
// MI455X (gfx1250) — matrix operand order repaired and checked by reading, not compiled, not run on hardware
//
#include <hip/hip_runtime.h>

#define Bsz 2
#define Ssz 2048
#define Dsz 1024
#define Hsz 16
#define Fsz 4096
#define Lsz 2
#define HDsz 64
#define MR (Bsz * Ssz)   // 4096 token rows

typedef __attribute__((ext_vector_type(16))) __bf16 v16bf;
typedef __attribute__((ext_vector_type(8)))  float  v8f;
typedef __bf16 bf16;

#define LOG2E 1.4426950408889634f

union V16 {
  v16bf v;
  uint32_t u[8];
};

// ---------------------------------------------------------------- WMMA helpers

__device__ __forceinline__ v8f wmma_bf16(v16bf a, v16bf b, v8f c) {
  // D = A(16x32 bf16) * B(32x16 bf16) + C(16x16 f32)
  return __builtin_amdgcn_wmma_f32_16x16x32_bf16(false, a, false, b, (short)0, c,
                                                 false, false);
}

__device__ __forceinline__ v8f zero_v8f() {
  v8f z;
#pragma unroll
  for (int i = 0; i < 8; ++i) z[i] = 0.f;
  return z;
}

// ---- gfx1250 async global->LDS copy (ASYNCcnt path), ISA §15.18 op 98 ----
__device__ __forceinline__ void async_copy_b128(const bf16* gsrc, bf16* ldst) {
  const unsigned long long ga = (unsigned long long)(uintptr_t)gsrc;
  const unsigned int la = (unsigned int)(uintptr_t)ldst;  // LDS offset = addr[31:0]
  asm volatile("global_load_async_to_lds_b128 %0, %1, off"
               :
               : "v"(la), "v"(ga)
               : "memory");
}
__device__ __forceinline__ void wait_async() {
  asm volatile("s_wait_asynccnt 0x0" ::: "memory");
}

// A-matrix 16x32 bf16 fragment from a row-major LDS tile (t -> (0,0), ld elems).
// ISA layout: lanes 0-15 & 16-31 both hold rows M=lane&15; K pairs:
//  lanes 0-15 : VGPR0..7 -> K={0,1},{2,3},{4,5},{6,7},{16,17},{18,19},{20,21},{22,23}
//  lanes16-31 : same + 8.  Each pair is contiguous -> b32/b128 LDS loads.
__device__ __forceinline__ v16bf load_a_frag(const bf16* t, int ld) {
  const int lane = threadIdx.x & 31;
  const int half = lane >> 4;
  const int r = lane & 15;
  V16 x;
#pragma unroll
  for (int j = 0; j < 8; ++j) {
    const int k = ((j < 4) ? (2 * j) : (2 * j + 8)) + 8 * half;
    x.u[j] = *(const uint32_t*)(t + r * ld + k);
  }
  return x.v;
}

// B-matrix 32x16 bf16 fragment, gathering from a tile stored transposed:
// logical B[k][n] = t[n*ld + k]  (same K pairs as A: lanes 0-15: K=0..7,16..23,
// lanes16-31: same + 8; N = lane&15 -> contiguous loads).
__device__ __forceinline__ v16bf load_bt_frag(const bf16* t, int ld) {
  const int lane = threadIdx.x & 31;
  const int n = lane & 15;
  const int kb = (lane >> 4) * 8;
  V16 x;
#pragma unroll
  for (int j = 0; j < 8; ++j) {
    const int k = kb + ((j < 4) ? (2 * j) : (2 * j + 8));
    x.u[j] = *(const uint32_t*)(t + n * ld + k);
  }
  return x.v;
}

// GELU (tanh approx) via native exp2/rcp: tanh(y) = sign(y)*(1 - 2/(2^(2*log2e*|y|)+1))
__device__ __forceinline__ float gelu_tanh(float x) {
  const float y = 0.7978845608f * (x + 0.044715f * x * x * x);
  const float ay = fabsf(y);
  const float t = __builtin_amdgcn_exp2f(2.8853900818f * ay);  // 2*log2(e)*|y|
  const float r = 1.f - 2.f * __builtin_amdgcn_rcpf(t + 1.f);
  return 0.5f * x * (1.f + copysignf(r, y));
}

// ---------------------------------------------------------------- small kernels

__global__ __launch_bounds__(256) void embed_kernel(const int* __restrict__ ids,
                                                    const float* __restrict__ we,
                                                    const float* __restrict__ pe,
                                                    float* __restrict__ X) {
  const int idx = blockIdx.x * 256 + threadIdx.x;     // over MR*Dsz
  const int bs = idx >> 10;
  const int d = idx & (Dsz - 1);
  const int s = bs & (Ssz - 1);
  X[idx] = we[(size_t)ids[bs] * Dsz + d] + pe[s * Dsz + d];
}

__global__ __launch_bounds__(256) void f32_to_bf16_kernel(const float* __restrict__ in,
                                                          bf16* __restrict__ out, int n) {
  const int i = blockIdx.x * 256 + threadIdx.x;
  if (i < n) out[i] = (bf16)in[i];
}

// One block per row of D=1024. obf!=null -> bf16 out; else f32*maskrow out.
__global__ __launch_bounds__(256) void layernorm_kernel(const float* __restrict__ X,
                                                        const float* __restrict__ w,
                                                        const float* __restrict__ bpar,
                                                        const float* __restrict__ mask,
                                                        bf16* __restrict__ obf,
                                                        float* __restrict__ of) {
  const int row = blockIdx.x;
  const float* x = X + (size_t)row * Dsz;
  float v[4];
  float s1 = 0.f, s2 = 0.f;
#pragma unroll
  for (int j = 0; j < 4; ++j) {
    v[j] = x[threadIdx.x + 256 * j];
    s1 += v[j];
    s2 += v[j] * v[j];
  }
#pragma unroll
  for (int o = 16; o > 0; o >>= 1) {
    s1 += __shfl_down(s1, o, 32);
    s2 += __shfl_down(s2, o, 32);
  }
  __shared__ float red[16];
  const int wave = threadIdx.x >> 5, lane = threadIdx.x & 31;
  if (lane == 0) { red[wave] = s1; red[8 + wave] = s2; }
  __syncthreads();
  if (threadIdx.x == 0) {
    float a = 0.f, b = 0.f;
#pragma unroll
    for (int i = 0; i < 8; ++i) { a += red[i]; b += red[8 + i]; }
    red[0] = a;
    red[8] = b;
  }
  __syncthreads();
  const float mean = red[0] * (1.f / Dsz);
  const float var = red[8] * (1.f / Dsz) - mean * mean;
  const float rstd = rsqrtf(var + 1e-5f);
  const float mrow = mask ? mask[row] : 1.f;
#pragma unroll
  for (int j = 0; j < 4; ++j) {
    const int d = threadIdx.x + 256 * j;
    const float y = (v[j] - mean) * rstd * w[d] + bpar[d];
    if (obf) obf[(size_t)row * Dsz + d] = (bf16)y;
    else     of[(size_t)row * Dsz + d] = y * mrow;
  }
}

__global__ __launch_bounds__(256) void qkv_split_kernel(const bf16* __restrict__ qkv,
                                                        bf16* __restrict__ q,
                                                        bf16* __restrict__ k,
                                                        bf16* __restrict__ v) {
  const int idx = blockIdx.x * 256 + threadIdx.x;     // over MR*Dsz
  const int bs = idx >> 10;
  const int d = idx & (Dsz - 1);
  const int b = bs >> 11;                 // /Ssz
  const int s = bs & (Ssz - 1);
  const int h = d >> 6;
  const int hd = d & 63;
  const size_t src = (size_t)bs * (3 * Dsz) + d;
  const size_t dst = (((size_t)(b * Hsz + h)) * Ssz + s) * HDsz + hd;
  q[dst] = qkv[src];
  k[dst] = qkv[src + Dsz];
  v[dst] = qkv[src + 2 * Dsz];
}

// ---------------------------------------------------------------- WMMA GEMM
// C[M,N] = A[M,K](bf16) x W[K,N](bf16) + bias, epilogue by template:
//   EPI==1 : bf16 out (QKV)
//   EPI==2 : bf16 out with GELU (MLP up)
//   EPI==3 : f32 out with residual add (proj / MLP down)
// 256 threads = 8 waves; WG tile 128x128; wave tile 64x32 (4x2 WMMA); K step 32.
// A tile staged via async global->LDS copies; weight tile staged *transposed*
// through VGPRs so B-fragment pairs are contiguous.
template <int EPI>
__global__ __launch_bounds__(256) void gemm_bf16_kernel(
    const bf16* __restrict__ A, const bf16* __restrict__ W,
    const float* __restrict__ bias, const float* __restrict__ resid,
    float* __restrict__ Cf, bf16* __restrict__ Cb, int M, int N, int K) {
  __shared__ alignas(16) bf16 sA[128 * 32];
  __shared__ alignas(16) bf16 sBT[128 * 32];   // [n][k] transposed
  const int tid = threadIdx.x;
  const int m0 = blockIdx.y * 128;
  const int n0 = blockIdx.x * 128;
  const int wave = tid >> 5;
  const int wr = wave & 1;   // 64-row group
  const int wc = wave >> 1;  // 32-col group (0..3)

  v8f acc[4][2];
#pragma unroll
  for (int mi = 0; mi < 4; ++mi)
#pragma unroll
    for (int ni = 0; ni < 2; ++ni) acc[mi][ni] = zero_v8f();

  const int ar = tid >> 1, ac = (tid & 1) * 16;
  const int br = tid & 31, bc = (tid >> 5) * 16;   // W: k row br, n chunk bc

  for (int k0 = 0; k0 < K; k0 += 32) {
    {
      // A tile: async global->LDS (no VGPR bounce), 2 x b128 per thread
      const bf16* g = A + (size_t)(m0 + ar) * K + k0 + ac;
      bf16* s = sA + ar * 32 + ac;
      async_copy_b128(g, s);
      async_copy_b128(g + 8, s + 8);
    }
    {
      // row br of W (k = k0+br), 16 consecutive n -> transpose into sBT[n][k]
      uint4 wv4[2];
      wv4[0] = *(const uint4*)(W + (size_t)(k0 + br) * N + n0 + bc);
      wv4[1] = *(const uint4*)(W + (size_t)(k0 + br) * N + n0 + bc + 8);
      const bf16* wv = (const bf16*)wv4;
#pragma unroll
      for (int i = 0; i < 16; ++i) sBT[(bc + i) * 32 + br] = wv[i];
    }
    if (k0 + 32 < K) {  // gfx1250 global_prefetch of next K-step
      __builtin_prefetch(A + (size_t)(m0 + ar) * K + k0 + 32 + ac, 0, 1);
      __builtin_prefetch(W + (size_t)(k0 + 32 + br) * N + n0 + bc, 0, 1);
    }
    wait_async();
    __syncthreads();

    v16bf a[4], bfr[2];
#pragma unroll
    for (int mi = 0; mi < 4; ++mi)
      a[mi] = load_a_frag(sA + (64 * wr + 16 * mi) * 32, 32);
#pragma unroll
    for (int ni = 0; ni < 2; ++ni)
      bfr[ni] = load_bt_frag(sBT + (32 * wc + 16 * ni) * 32, 32);
#pragma unroll
    for (int mi = 0; mi < 4; ++mi)
#pragma unroll
      for (int ni = 0; ni < 2; ++ni)
        acc[mi][ni] = wmma_bf16(a[mi], bfr[ni], acc[mi][ni]);
    __syncthreads();
  }

  const int lane = tid & 31;
  const int nl = lane & 15;
  const int mo = (lane >> 4) * 8;   // C/D rows: VGPR i -> row i (+8 for hi half)
#pragma unroll
  for (int mi = 0; mi < 4; ++mi)
#pragma unroll
    for (int ni = 0; ni < 2; ++ni) {
      const int rbase = m0 + 64 * wr + 16 * mi + mo;
      const int col = n0 + 32 * wc + 16 * ni + nl;
      const float bv = bias[col];
#pragma unroll
      for (int i = 0; i < 8; ++i) {
        const int row = rbase + i;
        float x = acc[mi][ni][i] + bv;
        if (EPI == 2) x = gelu_tanh(x);
        if (EPI == 3) x += resid[(size_t)row * N + col];
        if (EPI == 3) Cf[(size_t)row * N + col] = x;
        else          Cb[(size_t)row * N + col] = (bf16)x;
      }
    }
}

// ---------------------------------------------------------------- flash attention
// Q/K/V: bf16 [B*H][S][64]. grid = (S/64, B*H), block = 128 (4 waves).
// Each wave owns 16 queries; loop over 32-key blocks with online softmax.
// Softmax runs in the log2 domain so every exponential is one v_exp_f32.
// Q and K tiles staged with async global->LDS copies.
__global__ __launch_bounds__(128) void flash_attn_kernel(
    const bf16* __restrict__ Q, const bf16* __restrict__ K,
    const bf16* __restrict__ V, const float* __restrict__ mask,
    float* __restrict__ O) {
  __shared__ alignas(16) bf16 sQ[64 * 64];
  __shared__ alignas(16) bf16 sK[32 * 64];    // [key][hd] (row-major; OK for Q*K^T)
  __shared__ alignas(16) bf16 sVT[64 * 32];   // [hd][key] transposed for PV
  __shared__ alignas(16) bf16 sP[4][16 * 32];
  __shared__ float sNeg[32];

  const int tid = threadIdx.x;
  const int wave = tid >> 5, lane = tid & 31;
  const int bh = blockIdx.y;
  const int b = bh >> 4, h = bh & 15;
  const int q0 = blockIdx.x * 64;

  {  // stage 64x64 Q block via async copies (4 x b128 per thread)
    const bf16* gQ = Q + ((size_t)bh * Ssz + q0) * HDsz;
    const int r = tid >> 1, c = (tid & 1) * 32;
    const bf16* g = gQ + r * HDsz + c;
    bf16* s = sQ + r * HDsz + c;
    async_copy_b128(g, s);
    async_copy_b128(g + 8, s + 8);
    async_copy_b128(g + 16, s + 16);
    async_copy_b128(g + 24, s + 24);
  }
  wait_async();
  __syncthreads();
  const v16bf aq0 = load_a_frag(sQ + (16 * wave) * HDsz, HDsz);        // hd 0..31
  const v16bf aq1 = load_a_frag(sQ + (16 * wave) * HDsz + 32, HDsz);   // hd 32..63

  float m_s[8], l_s[8];
#pragma unroll
  for (int i = 0; i < 8; ++i) { m_s[i] = -1e30f; l_s[i] = 0.f; }
  v8f o[4];
#pragma unroll
  for (int c = 0; c < 4; ++c) o[c] = zero_v8f();

  const float scale2 = 0.125f * LOG2E;  // 1/sqrt(64) in log2 domain
  bf16* sPw = sP[wave];

  for (int kb2 = 0; kb2 < Ssz / 32; ++kb2) {
    __syncthreads();  // previous PV done before overwriting sK/sVT
    {
      const bf16* gK = K + ((size_t)bh * Ssz + kb2 * 32) * HDsz;
      const bf16* gV = V + ((size_t)bh * Ssz + kb2 * 32) * HDsz;
      const int r = tid >> 2, c = (tid & 3) * 16;   // key row r, hd chunk c
      // K tile: row-major copy -> async global->LDS (2 x b128)
      async_copy_b128(gK + r * HDsz + c, sK + r * HDsz + c);
      async_copy_b128(gK + r * HDsz + c + 8, sK + r * HDsz + c + 8);
      // V tile: transposed staging (needs VGPR bounce)
      uint4 vv4[2];
      vv4[0] = *(const uint4*)(gV + r * HDsz + c);
      vv4[1] = *(const uint4*)(gV + r * HDsz + c + 8);
      const bf16* vv = (const bf16*)vv4;
#pragma unroll
      for (int i = 0; i < 16; ++i) sVT[(c + i) * 32 + r] = vv[i];  // transpose
    }
    if (tid < 32)
      sNeg[tid] = (1.f - mask[b * Ssz + kb2 * 32 + tid]) * (-1e9f * LOG2E);
    wait_async();
    __syncthreads();

    // scores S[16 queries x 32 keys] = Q * K^T  (4 WMMA)
    // B[k=hd][n=key] = sK[key*64 + hd] -> load_bt_frag on row-major sK.
    v8f s0 = zero_v8f(), s1 = zero_v8f();
    s0 = wmma_bf16(aq0, load_bt_frag(sK, HDsz), s0);
    s0 = wmma_bf16(aq1, load_bt_frag(sK + 32, HDsz), s0);
    s1 = wmma_bf16(aq0, load_bt_frag(sK + 16 * HDsz, HDsz), s1);
    s1 = wmma_bf16(aq1, load_bt_frag(sK + 16 * HDsz + 32, HDsz), s1);

    const float neg0 = sNeg[lane & 15];
    const float neg1 = sNeg[16 + (lane & 15)];
    float al[8];
#pragma unroll
    for (int i = 0; i < 8; ++i) {
      float v0 = s0[i] * scale2 + neg0;   // log2-domain score
      float v1 = s1[i] * scale2 + neg1;
      float mx = fmaxf(v0, v1);
#pragma unroll
      for (int off = 1; off < 16; off <<= 1) mx = fmaxf(mx, __shfl_xor(mx, off, 32));
      const float mnew = fmaxf(m_s[i], mx);
      al[i] = __builtin_amdgcn_exp2f(m_s[i] - mnew);
      const float p0 = __builtin_amdgcn_exp2f(v0 - mnew);
      const float p1 = __builtin_amdgcn_exp2f(v1 - mnew);
      float rs = p0 + p1;
#pragma unroll
      for (int off = 1; off < 16; off <<= 1) rs += __shfl_xor(rs, off, 32);
      l_s[i] = l_s[i] * al[i] + rs;
      m_s[i] = mnew;
      // C layout -> row-major P staging (rows i / i+8 per lane half)
      const int prow = (lane >> 4) * 8 + i;
      sPw[prow * 32 + (lane & 15)] = (bf16)p0;
      sPw[prow * 32 + 16 + (lane & 15)] = (bf16)p1;
    }
#pragma unroll
    for (int c = 0; c < 4; ++c)
#pragma unroll
      for (int i = 0; i < 8; ++i) o[c][i] = o[c][i] * al[i];
    __syncthreads();  // P visible (per-wave region; also keeps waves in step)

    // O += P(16x32) * V(32x64): B[k=key][n=hd] = sVT[hd*32 + key]  (4 WMMA)
    const v16bf ap = load_a_frag(sPw, 32);
#pragma unroll
    for (int c = 0; c < 4; ++c)
      o[c] = wmma_bf16(ap, load_bt_frag(sVT + (c * 16) * 32, 32), o[c]);
  }

  // write O (head-interleaved [B,S,D]) with 1/l normalization
#pragma unroll
  for (int c = 0; c < 4; ++c)
#pragma unroll
    for (int i = 0; i < 8; ++i) {
      const int q = q0 + 16 * wave + (lane >> 4) * 8 + i;
      const int d = c * 16 + (lane & 15);
      O[((size_t)(b * Ssz + q)) * Dsz + h * HDsz + d] = o[c][i] / l_s[i];
    }
}

// ---------------------------------------------------------------- driver

extern "C" void kernel_launch(void* const* d_in, const int* in_sizes, int n_in,
                              void* d_out, int out_size, void* d_ws, size_t ws_size,
                              hipStream_t stream) {
  const int*   ids  = (const int*)d_in[0];
  const float* mask = (const float*)d_in[1];
  const float* wemb = (const float*)d_in[2];
  const float* pemb = (const float*)d_in[3];
  const float* ln1w = (const float*)d_in[4];
  const float* ln1b = (const float*)d_in[5];
  const float* wqkv = (const float*)d_in[6];
  const float* bqkv = (const float*)d_in[7];
  const float* wo   = (const float*)d_in[8];
  const float* bo   = (const float*)d_in[9];
  const float* ln2w = (const float*)d_in[10];
  const float* ln2b = (const float*)d_in[11];
  const float* w1   = (const float*)d_in[12];
  const float* b1   = (const float*)d_in[13];
  const float* w2   = (const float*)d_in[14];
  const float* b2   = (const float*)d_in[15];
  const float* nfw  = (const float*)d_in[16];
  const float* nfb  = (const float*)d_in[17];

  char* p = (char*)d_ws;
  auto carve = [&](size_t bytes) -> void* {
    void* r = (void*)p;
    p += (bytes + 255) & ~(size_t)255;
    return r;
  };
  float* X     = (float*)carve((size_t)MR * Dsz * 4);
  bf16*  hbf   = (bf16*)carve((size_t)MR * Dsz * 2);
  bf16*  qkvbf = (bf16*)carve((size_t)MR * 3 * Dsz * 2);
  bf16*  qb    = (bf16*)carve((size_t)MR * Dsz * 2);
  bf16*  kbuf  = (bf16*)carve((size_t)MR * Dsz * 2);
  bf16*  vbuf  = (bf16*)carve((size_t)MR * Dsz * 2);
  float* Obuf  = (float*)carve((size_t)MR * Dsz * 4);
  bf16*  mlpbf = (bf16*)carve((size_t)MR * Fsz * 2);
  bf16*  wqkvbf = (bf16*)carve((size_t)Lsz * Dsz * 3 * Dsz * 2);
  bf16*  wobf   = (bf16*)carve((size_t)Lsz * Dsz * Dsz * 2);
  bf16*  w1bf   = (bf16*)carve((size_t)Lsz * Dsz * Fsz * 2);
  bf16*  w2bf   = (bf16*)carve((size_t)Lsz * Fsz * Dsz * 2);

  auto cvt = [&](const float* src, bf16* dst, int n) {
    f32_to_bf16_kernel<<<(n + 255) / 256, 256, 0, stream>>>(src, dst, n);
  };
  cvt(wqkv, wqkvbf, Lsz * Dsz * 3 * Dsz);
  cvt(wo, wobf, Lsz * Dsz * Dsz);
  cvt(w1, w1bf, Lsz * Dsz * Fsz);
  cvt(w2, w2bf, Lsz * Fsz * Dsz);

  embed_kernel<<<MR * Dsz / 256, 256, 0, stream>>>(ids, wemb, pemb, X);

  for (int l = 0; l < Lsz; ++l) {
    layernorm_kernel<<<MR, 256, 0, stream>>>(X, ln1w + l * Dsz, ln1b + l * Dsz,
                                             nullptr, hbf, nullptr);
    gemm_bf16_kernel<1><<<dim3(3 * Dsz / 128, MR / 128), 256, 0, stream>>>(
        hbf, wqkvbf + (size_t)l * Dsz * 3 * Dsz, bqkv + l * 3 * Dsz, nullptr,
        nullptr, qkvbf, MR, 3 * Dsz, Dsz);
    qkv_split_kernel<<<MR * Dsz / 256, 256, 0, stream>>>(qkvbf, qb, kbuf, vbuf);
    flash_attn_kernel<<<dim3(Ssz / 64, Bsz * Hsz), 128, 0, stream>>>(
        qb, kbuf, vbuf, mask, Obuf);
    cvt(Obuf, hbf, MR * Dsz);
    gemm_bf16_kernel<3><<<dim3(Dsz / 128, MR / 128), 256, 0, stream>>>(
        hbf, wobf + (size_t)l * Dsz * Dsz, bo + l * Dsz, X, X, nullptr,
        MR, Dsz, Dsz);
    layernorm_kernel<<<MR, 256, 0, stream>>>(X, ln2w + l * Dsz, ln2b + l * Dsz,
                                             nullptr, hbf, nullptr);
    gemm_bf16_kernel<2><<<dim3(Fsz / 128, MR / 128), 256, 0, stream>>>(
        hbf, w1bf + (size_t)l * Dsz * Fsz, b1 + l * Fsz, nullptr, nullptr, mlpbf,
        MR, Fsz, Dsz);
    gemm_bf16_kernel<3><<<dim3(Dsz / 128, MR / 128), 256, 0, stream>>>(
        mlpbf, w2bf + (size_t)l * Fsz * Dsz, b2 + l * Dsz, X, X, nullptr,
        MR, Dsz, Fsz);
  }
  layernorm_kernel<<<MR, 256, 0, stream>>>(X, nfw, nfb, mask, nullptr,
                                           (float*)d_out);
  (void)in_sizes; (void)n_in; (void)out_size; (void)ws_size;
}
